// MPNN_rbf_56581899157524
// MI455X (gfx1250) — matrix operand order repaired and checked by reading, not compiled, not run on hardware
//
#include <hip/hip_runtime.h>
#include <hip/hip_bf16.h>

// ---------------------------------------------------------------------------
// MPNN (NNConv + GRU + Set2Set) for MI455X / gfx1250.
// Strategy:
//   * Edge-network 2nd layer ([E,128]@[128,4096], 34.4 GFLOP) done ONCE with
//     v_wmma_f32_16x16x32_f16 and stored as f16 (256 MB) -> the 6 MP steps
//     become bandwidth-bound streaming passes (~75us floor @ 23.3 TB/s
//     for 1x write + 6x read of w_edge).
//   * WMMA fragments loaded straight from global (both operands are
//     K-contiguous 16B chunks per lane per ISA 7.12.2) -> no LDS staging.
//   * k_msg streams w_edge with global_load_b128 (512B/instr across the wave)
//     + shfl_xor cross-lane reduction; segment sums via global_atomic_add_f32.
// ---------------------------------------------------------------------------

#define N_NODES 16384
#define N_EDGES 32768
#define N_GRAPH 512
#define DN      64
#define DE      128
#define OUT_D   12
#define STEPS   6

typedef __attribute__((ext_vector_type(16))) _Float16 v16h;
typedef __attribute__((ext_vector_type(8)))  float    v8f;

__device__ __forceinline__ float sigf(float x) { return 1.0f / (1.0f + __expf(-x)); }

__device__ __forceinline__ void atomicMaxF(float* addr, float val) {
    unsigned int* a = (unsigned int*)addr;
    unsigned int old = *a;
    while (true) {
        float f = __uint_as_float(old);
        if (f >= val) break;
        unsigned int assumed = old;
        old = atomicCAS(a, assumed, __float_as_uint(val));
        if (old == assumed) break;
    }
}

// ---------------------------------------------------------------- fill -----
__global__ void k_fill(float* __restrict__ p, float v, int n) {
    int i = blockIdx.x * blockDim.x + threadIdx.x;
    if (i < n) p[i] = v;
}

// ------------------------------------------------------- lin0 + relu -------
// h[n, t] = relu(x[n,:15] . w_lin0[t,:15] + b[t]);  one 64-thread block/node
__global__ void k_lin0(const float* __restrict__ x, const float* __restrict__ w,
                       const float* __restrict__ b, float* __restrict__ h) {
    __shared__ float xs[15];
    int n = blockIdx.x, t = threadIdx.x;
    if (t < 15) xs[t] = x[(size_t)n * 15 + t];
    __syncthreads();
    float acc = b[t];
#pragma unroll
    for (int k = 0; k < 15; ++k) acc += xs[k] * w[t * 15 + k];
    h[(size_t)n * DN + t] = fmaxf(acc, 0.0f);
}

// --------------------------------------- RBF + edge-net layer1 + relu ------
// hid[e, t] = relu(e_feat[54] . w_en1[t,:54] + b[t]), stored f16.
// e_feat = [edge_attr[:,0:4], exp(-(d - c_i)^2 / g)], c = linspace(0,5,50)
__global__ void k_rbf_en1(const float* __restrict__ edge_attr,
                          const float* __restrict__ w_en1,
                          const float* __restrict__ b_en1,
                          _Float16* __restrict__ hid) {
    __shared__ float es[54];
    int e = blockIdx.x, t = threadIdx.x; // 128 threads
    float dist = edge_attr[(size_t)e * 5 + 4];
    if (t < 4) es[t] = edge_attr[(size_t)e * 5 + t];
    if (t < 50) {
        const float g = 5.0f / 49.0f;
        float r = dist - g * (float)t;
        es[4 + t] = __expf(-(r * r) / g);
    }
    __syncthreads();
    float acc = b_en1[t];
    const float* w = w_en1 + (size_t)t * 54;
#pragma unroll 6
    for (int k = 0; k < 54; ++k) acc += es[k] * w[k];
    hid[(size_t)e * DE + t] = (_Float16)fmaxf(acc, 0.0f);
}

// ---------------------------------------------------- f32 -> f16 convert ---
__global__ void k_cvt_f16(const float* __restrict__ src, _Float16* __restrict__ dst, int n) {
    int i = blockIdx.x * blockDim.x + threadIdx.x;
    if (i < n) dst[i] = (_Float16)src[i];
}

// ------------------------------------------- WMMA edge-weight GEMM ---------
// w_edge[E, 4096] = hid[E,128] @ w_en2^T[128,4096] + b_en2  (f16 in/out, f32 acc)
// Tile: wave computes a 16(edge) x 16(col) block, K-loop of 4 x (16x16x32).
// A frag: hid rows, K-contiguous.  B frag: w_en2 row j is K-contiguous.
__global__ __launch_bounds__(256)
void k_edge_gemm_wmma(const _Float16* __restrict__ hid,
                      const _Float16* __restrict__ w2f16,
                      const float* __restrict__ b_en2,
                      _Float16* __restrict__ w_edge) {
    const int lane = threadIdx.x & 31;
    const int wav  = threadIdx.x >> 5;                 // 0..7
    const int ebase = blockIdx.x * 16;                 // edge-row tile
    const int jbase = (blockIdx.y * 8 + wav) * 16;     // column tile in [0,4096)
    const int mr    = lane & 15;
    const int hs    = lane >> 4;                       // half-select

    // A: lane holds row (ebase+mr), halves K = k0 + 8*hs + {0..7} and +16
    const _Float16* aRow = hid   + (size_t)(ebase + mr) * DE + hs * 8;
    // B: lane holds col (jbase+mr), halves K = k0 + 8*hs + {0..7} and +16
    const _Float16* bCol = w2f16 + (size_t)(jbase + mr) * DE + hs * 8;

    v8f c = {};
#pragma unroll
    for (int k0 = 0; k0 < DE; k0 += 32) {
        union { v16h v; uint4 u[2]; } A, B;
        A.u[0] = *(const uint4*)(aRow + k0);
        A.u[1] = *(const uint4*)(aRow + k0 + 16);
        B.u[0] = *(const uint4*)(bCol + k0);
        B.u[1] = *(const uint4*)(bCol + k0 + 16);
        c = __builtin_amdgcn_wmma_f32_16x16x32_f16(false, A.v, false, B.v,
                                                   (short)0, c, false, false);
    }
    // D layout: VGPR r -> M = r + 8*hs, N = lane&15
    const int nc = lane & 15;
    const float bias = b_en2[jbase + nc];
#pragma unroll
    for (int r = 0; r < 8; ++r) {
        int mg = ebase + r + 8 * hs;
        w_edge[(size_t)mg * 4096 + jbase + nc] = (_Float16)(c[r] + bias);
    }
}

// ----------------------------------------------------------- degree --------
__global__ void k_deg(const int* __restrict__ dst, float* __restrict__ deg, int E) {
    int i = blockIdx.x * blockDim.x + threadIdx.x;
    if (i < E) atomicAdd(&deg[dst[i]], 1.0f);
}

// ------------------------------------------- message + scatter-add ---------
// One wave per edge: msg = h[src] (1x64) * W_e (64x64, f16), atomicAdd to
// agg[dst].  Each lane streams contiguous uint4 (8 halves) so a wave moves
// 512B per VMEM instruction (rows in groups of 4: lane = 4 rows x 8-col grp).
// Cross-lane: lanes {l, l^8, l^16, l^24} share a column group -> shfl_xor.
__global__ __launch_bounds__(256)
void k_msg(const float* __restrict__ h, const _Float16* __restrict__ w_edge,
           const int* __restrict__ src, const int* __restrict__ dst,
           float* __restrict__ agg) {
    int wave = (blockIdx.x * blockDim.x + threadIdx.x) >> 5;
    int lane = threadIdx.x & 31;
    if (wave >= N_EDGES) return;
    int s = src[wave], d = dst[wave];
    const float* hr = h + (size_t)s * DN;
    float h_lo = hr[lane];
    float h_hi = hr[lane + 32];

    const int cg   = (lane & 7) * 8;   // this lane's 8-column group base
    const int rsub = lane >> 3;        // row-within-4-row-block (0..3)
    const _Float16* W = w_edge + (size_t)wave * 4096 + cg;

    float facc[8] = {0.f, 0.f, 0.f, 0.f, 0.f, 0.f, 0.f, 0.f};
#pragma unroll 4
    for (int db = 0; db < DN; db += 4) {
        int r = db + rsub;
        float hval = (db < 32) ? __shfl(h_lo, r, 32) : __shfl(h_hi, r - 32, 32);
        union { uint4 u; _Float16 hx[8]; } w;
        w.u = *(const uint4*)(W + (size_t)r * DN);   // global_load_b128
#pragma unroll
        for (int j = 0; j < 8; ++j) facc[j] += hval * (float)w.hx[j];
    }
#pragma unroll
    for (int j = 0; j < 8; ++j) {
        facc[j] += __shfl_xor(facc[j], 8, 32);
        facc[j] += __shfl_xor(facc[j], 16, 32);
    }
    // the 4 lanes of each column group each commit 2 of its 8 columns
    int j0 = rsub * 2;
    atomicAdd(&agg[(size_t)d * DN + cg + j0],     facc[j0]);
    atomicAdd(&agg[(size_t)d * DN + cg + j0 + 1], facc[j0 + 1]);
}

// ------------------------------------------------- GRU cell (per node) -----
// m = relu(agg/deg + b_conv); h = GRU(m, h). 64-thread block per node.
__global__ void k_gru(const float* __restrict__ agg, const float* __restrict__ deg,
                      const float* __restrict__ b_conv,
                      const float* __restrict__ w_ih, const float* __restrict__ w_hh,
                      const float* __restrict__ b_ih, const float* __restrict__ b_hh,
                      float* __restrict__ h) {
    __shared__ float ms[DN], hsv[DN];
    int n = blockIdx.x, t = threadIdx.x;
    float dg = fmaxf(deg[n], 1.0f);
    float m = fmaxf(agg[(size_t)n * DN + t] / dg + b_conv[t], 0.0f);
    ms[t] = m;
    hsv[t] = h[(size_t)n * DN + t];
    __syncthreads();
    float gi_r = b_ih[t], gi_z = b_ih[DN + t], gi_n = b_ih[2 * DN + t];
    float gh_r = b_hh[t], gh_z = b_hh[DN + t], gh_n = b_hh[2 * DN + t];
#pragma unroll 4
    for (int d = 0; d < DN; ++d) {
        float mv = ms[d], hv = hsv[d];
        gi_r += mv * w_ih[(t) * DN + d];
        gi_z += mv * w_ih[(DN + t) * DN + d];
        gi_n += mv * w_ih[(2 * DN + t) * DN + d];
        gh_r += hv * w_hh[(t) * DN + d];
        gh_z += hv * w_hh[(DN + t) * DN + d];
        gh_n += hv * w_hh[(2 * DN + t) * DN + d];
    }
    float r  = sigf(gi_r + gh_r);
    float z  = sigf(gi_z + gh_z);
    float nn = tanhf(gi_n + r * gh_n);
    h[(size_t)n * DN + t] = (1.0f - z) * nn + z * hsv[t];
}

// ------------------------------------------- Set2Set LSTM step -------------
// 256-thread block per graph: gates = qstar@Wih^T + hl@Whh^T + biases.
__global__ void k_lstm(const float* __restrict__ qstar,
                       const float* __restrict__ w_ih_l, const float* __restrict__ w_hh_l,
                       const float* __restrict__ b_ih_l, const float* __restrict__ b_hh_l,
                       float* __restrict__ hl, float* __restrict__ cl,
                       float* __restrict__ q) {
    __shared__ float gates[4 * DN];
    __shared__ float qs[2 * DN];
    __shared__ float hs[DN];
    int g = blockIdx.x, t = threadIdx.x; // 256 threads
    if (t < 2 * DN) qs[t] = qstar[(size_t)g * 2 * DN + t];
    if (t < DN)     hs[t] = hl[(size_t)g * DN + t];
    __syncthreads();
    float acc = b_ih_l[t] + b_hh_l[t];
    const float* wi = w_ih_l + (size_t)t * 2 * DN;
    const float* wh = w_hh_l + (size_t)t * DN;
#pragma unroll 4
    for (int k = 0; k < 2 * DN; ++k) acc += qs[k] * wi[k];
#pragma unroll 4
    for (int k = 0; k < DN; ++k)     acc += hs[k] * wh[k];
    gates[t] = acc;
    __syncthreads();
    if (t < DN) {
        float ii = gates[t], ff = gates[DN + t], gg = gates[2 * DN + t], oo = gates[3 * DN + t];
        float c = sigf(ff) * cl[(size_t)g * DN + t] + sigf(ii) * tanhf(gg);
        float hh = sigf(oo) * tanhf(c);
        cl[(size_t)g * DN + t] = c;
        hl[(size_t)g * DN + t] = hh;
        q[(size_t)g * DN + t] = hh;
    }
}

// --------------------------------- attention energy + per-graph max --------
// One wave per node: energy[n] = h[n,:] . q[batch[n],:]
__global__ __launch_bounds__(256)
void k_energy(const float* __restrict__ h, const float* __restrict__ q,
              const int* __restrict__ batch, float* __restrict__ energy,
              float* __restrict__ emax) {
    int wave = (blockIdx.x * blockDim.x + threadIdx.x) >> 5;
    int lane = threadIdx.x & 31;
    if (wave >= N_NODES) return;
    int g = batch[wave];
    const float* hr = h + (size_t)wave * DN;
    const float* qr = q + (size_t)g * DN;
    float p = hr[lane] * qr[lane] + hr[lane + 32] * qr[lane + 32];
#pragma unroll
    for (int off = 16; off > 0; off >>= 1) p += __shfl_down(p, off, 32);
    if (lane == 0) {
        energy[wave] = p;
        atomicMaxF(&emax[g], p);
    }
}

// --------------------------------------- exp + per-graph sum ---------------
__global__ void k_ez(const float* __restrict__ energy, const float* __restrict__ emax,
                     const int* __restrict__ batch, float* __restrict__ ez,
                     float* __restrict__ esum) {
    int i = blockIdx.x * blockDim.x + threadIdx.x;
    if (i >= N_NODES) return;
    int g = batch[i];
    float v = __expf(energy[i] - emax[g]);
    ez[i] = v;
    atomicAdd(&esum[g], v);
}

// --------------------------------------- weighted readout sum --------------
__global__ __launch_bounds__(256)
void k_rread(const float* __restrict__ h, const float* __restrict__ ez,
             const float* __restrict__ esum, const int* __restrict__ batch,
             float* __restrict__ rread) {
    int wave = (blockIdx.x * blockDim.x + threadIdx.x) >> 5;
    int lane = threadIdx.x & 31;
    if (wave >= N_NODES) return;
    int g = batch[wave];
    float a = ez[wave] / esum[g];
    atomicAdd(&rread[(size_t)g * DN + lane],      a * h[(size_t)wave * DN + lane]);
    atomicAdd(&rread[(size_t)g * DN + lane + 32], a * h[(size_t)wave * DN + lane + 32]);
}

// --------------------------------------------- q_star = [q, r_read] --------
__global__ void k_qstar(const float* __restrict__ q, const float* __restrict__ rread,
                        float* __restrict__ qstar) {
    int i = blockIdx.x * blockDim.x + threadIdx.x; // G*DN
    if (i >= N_GRAPH * DN) return;
    int g = i / DN, t = i % DN;
    qstar[(size_t)g * 2 * DN + t]      = q[i];
    qstar[(size_t)g * 2 * DN + DN + t] = rread[i];
}

// ----------------------------------------------------- readout head --------
__global__ void k_head(const float* __restrict__ qstar,
                       const float* __restrict__ w1, const float* __restrict__ b1,
                       const float* __restrict__ w2, const float* __restrict__ b2,
                       float* __restrict__ out) {
    __shared__ float qs[2 * DN];
    __shared__ float y1[DN];
    int g = blockIdx.x, t = threadIdx.x; // 128 threads
    qs[t] = qstar[(size_t)g * 2 * DN + t];
    __syncthreads();
    if (t < DN) {
        float acc = b1[t];
#pragma unroll 4
        for (int k = 0; k < 2 * DN; ++k) acc += qs[k] * w1[t * 2 * DN + k];
        y1[t] = fmaxf(acc, 0.0f);
    }
    __syncthreads();
    if (t < OUT_D) {
        float acc = b2[t];
#pragma unroll 4
        for (int k = 0; k < DN; ++k) acc += y1[k] * w2[t * DN + k];
        out[(size_t)g * OUT_D + t] = acc;
    }
}

// ===========================================================================
extern "C" void kernel_launch(void* const* d_in, const int* in_sizes, int n_in,
                              void* d_out, int out_size, void* d_ws, size_t ws_size,
                              hipStream_t stream) {
    const float* x         = (const float*)d_in[0];
    const float* edge_attr = (const float*)d_in[1];
    const int*   edge_idx  = (const int*)  d_in[2];
    const int*   batch     = (const int*)  d_in[3];
    const float* w_lin0    = (const float*)d_in[4];
    const float* b_lin0    = (const float*)d_in[5];
    const float* w_en1     = (const float*)d_in[6];
    const float* b_en1     = (const float*)d_in[7];
    const float* w_en2     = (const float*)d_in[8];
    const float* b_en2     = (const float*)d_in[9];
    const float* b_conv    = (const float*)d_in[10];
    const float* w_ih      = (const float*)d_in[11];
    const float* w_hh      = (const float*)d_in[12];
    const float* b_ih      = (const float*)d_in[13];
    const float* b_hh      = (const float*)d_in[14];
    const float* w_ih_l    = (const float*)d_in[15];
    const float* w_hh_l    = (const float*)d_in[16];
    const float* b_ih_l    = (const float*)d_in[17];
    const float* b_hh_l    = (const float*)d_in[18];
    const float* w_lin1    = (const float*)d_in[19];
    const float* b_lin1    = (const float*)d_in[20];
    const float* w_lin2    = (const float*)d_in[21];
    const float* b_lin2    = (const float*)d_in[22];
    float* out = (float*)d_out;

    const int* src = edge_idx;           // edge_index[0,:]
    const int* dst = edge_idx + N_EDGES; // edge_index[1,:]

    // ---- workspace carve-up (256B aligned) ----
    char* wsp = (char*)d_ws;
    auto carve = [&](size_t bytes) {
        char* p = wsp;
        wsp += (bytes + 255) & ~(size_t)255;
        return p;
    };
    float*    h      = (float*)   carve((size_t)N_NODES * DN * 4);
    float*    agg    = (float*)   carve((size_t)N_NODES * DN * 4);
    float*    deg    = (float*)   carve((size_t)N_NODES * 4);
    _Float16* hid    = (_Float16*)carve((size_t)N_EDGES * DE * 2);
    _Float16* w2f16  = (_Float16*)carve((size_t)4096 * DE * 2);
    float*    qstar  = (float*)   carve((size_t)N_GRAPH * 2 * DN * 4);
    float*    hl     = (float*)   carve((size_t)N_GRAPH * DN * 4);
    float*    cl     = (float*)   carve((size_t)N_GRAPH * DN * 4);
    float*    q      = (float*)   carve((size_t)N_GRAPH * DN * 4);
    float*    rread  = (float*)   carve((size_t)N_GRAPH * DN * 4);
    float*    energy = (float*)   carve((size_t)N_NODES * 4);
    float*    ez     = (float*)   carve((size_t)N_NODES * 4);
    float*    emax   = (float*)   carve((size_t)N_GRAPH * 4);
    float*    esum   = (float*)   carve((size_t)N_GRAPH * 4);
    _Float16* w_edge = (_Float16*)carve((size_t)N_EDGES * 4096 * 2); // 256 MB, last

    auto blocks = [](int n, int t) { return (n + t - 1) / t; };

    // ---- one-time (per call) preprocessing ----
    k_lin0<<<N_NODES, DN, 0, stream>>>(x, w_lin0, b_lin0, h);
    k_rbf_en1<<<N_EDGES, DE, 0, stream>>>(edge_attr, w_en1, b_en1, hid);
    k_cvt_f16<<<blocks(4096 * DE, 256), 256, 0, stream>>>(w_en2, w2f16, 4096 * DE);
    {
        dim3 grid(N_EDGES / 16, 4096 / 128); // 8 waves/block * 16 cols = 128 cols
        k_edge_gemm_wmma<<<grid, 256, 0, stream>>>(hid, w2f16, b_en2, w_edge);
    }
    k_fill<<<blocks(N_NODES, 256), 256, 0, stream>>>(deg, 0.0f, N_NODES);
    k_deg<<<blocks(N_EDGES, 256), 256, 0, stream>>>(dst, deg, N_EDGES);

    // ---- message passing ----
    for (int s = 0; s < STEPS; ++s) {
        k_fill<<<blocks(N_NODES * DN, 256), 256, 0, stream>>>(agg, 0.0f, N_NODES * DN);
        k_msg<<<blocks(N_EDGES * 32, 256), 256, 0, stream>>>(h, w_edge, src, dst, agg);
        k_gru<<<N_NODES, DN, 0, stream>>>(agg, deg, b_conv, w_ih, w_hh, b_ih, b_hh, h);
    }

    // ---- Set2Set ----
    k_fill<<<blocks(N_GRAPH * 2 * DN, 256), 256, 0, stream>>>(qstar, 0.0f, N_GRAPH * 2 * DN);
    k_fill<<<blocks(N_GRAPH * DN, 256), 256, 0, stream>>>(hl, 0.0f, N_GRAPH * DN);
    k_fill<<<blocks(N_GRAPH * DN, 256), 256, 0, stream>>>(cl, 0.0f, N_GRAPH * DN);
    for (int s = 0; s < STEPS; ++s) {
        k_fill<<<blocks(N_GRAPH, 256), 256, 0, stream>>>(emax, -3.0e38f, N_GRAPH);
        k_fill<<<blocks(N_GRAPH, 256), 256, 0, stream>>>(esum, 0.0f, N_GRAPH);
        k_fill<<<blocks(N_GRAPH * DN, 256), 256, 0, stream>>>(rread, 0.0f, N_GRAPH * DN);
        k_lstm<<<N_GRAPH, 4 * DN, 0, stream>>>(qstar, w_ih_l, w_hh_l, b_ih_l, b_hh_l, hl, cl, q);
        k_energy<<<blocks(N_NODES * 32, 256), 256, 0, stream>>>(h, q, batch, energy, emax);
        k_ez<<<blocks(N_NODES, 256), 256, 0, stream>>>(energy, emax, batch, ez, esum);
        k_rread<<<blocks(N_NODES * 32, 256), 256, 0, stream>>>(h, ez, esum, batch, rread);
        k_qstar<<<blocks(N_GRAPH * DN, 256), 256, 0, stream>>>(q, rread, qstar);
    }

    // ---- head ----
    k_head<<<N_GRAPH, 2 * DN, 0, stream>>>(qstar, w_lin1, b_lin1, w_lin2, b_lin2, out);
}
